// Encoder_68341519614429
// MI455X (gfx1250) — matrix operand order repaired and checked by reading, not compiled, not run on hardware
//
#include <hip/hip_runtime.h>
#include <hip/hip_bf16.h>
#include <stdint.h>

// ---------------------------------------------------------------------------
// CDNA5 (gfx1250) transformer encoder block.
// All GEMMs + attention use V_WMMA_F32_16X16X32_F16 (wave32 WMMA).
// Attention K-tiles staged with GLOBAL_LOAD_ASYNC_TO_LDS_B128 (ASYNCcnt).
// ---------------------------------------------------------------------------

typedef __attribute__((ext_vector_type(16))) _Float16 v16h;
typedef __attribute__((ext_vector_type(8)))  _Float16 v8h;
typedef __attribute__((ext_vector_type(8)))  float    v8f;

#define D_MODEL 1024
#define HID     4096
#define NHEAD   16
#define DH      64
#define SEQ     2048
#define BATCH   2
#define ROWS    (BATCH * SEQ)      // 4096 token rows
#define THREE_D (3 * D_MODEL)      // 3072

#define USE_ASYNC_LDS 1            // global_load_async_to_lds_b128 staging

enum { EP_F16 = 0, EP_GELU = 1, EP_RESF32 = 2 };

__device__ __forceinline__ v8f wmma16x16x32(v16h a, v16h b, v8f c) {
  // D = A(16x32 f16) * B(32x16 f16) + C(16x16 f32)
  return __builtin_amdgcn_wmma_f32_16x16x32_f16(false, a, false, b,
                                                (short)0, c, false, false);
}

// A-fragment loader, contiguous k order (not the ISA table order):
// lane r (=lane&15) holds row M=r; lanes 0-15 get K = 0..15,
// lanes 16-31 get K = 16..31 -> two contiguous 16B chunks.
__device__ __forceinline__ v16h load_a16x32(const _Float16* A, int lda, int lane) {
  const int r  = lane & 15;
  const int hh = (lane >> 4) & 1;
  const _Float16* p = A + (size_t)r * lda + 16 * hh;
  v8h lo = *(const v8h*)(p);
  v8h hi = *(const v8h*)(p + 8);
  v16h f;
#pragma unroll
  for (int i = 0; i < 8; ++i) { f[i] = lo[i]; f[i + 8] = hi[i]; }
  return f;
}

// ---------------------------------------------------------------------------
// Weight transpose + f32 -> f16 cast: dst[n*K + k] = (f16) src[k*N + n].
// Makes WMMA B-fragments contiguous 32B loads (lane = column, 16 K-halfs).
// ---------------------------------------------------------------------------
__global__ __launch_bounds__(256) void transpose_cast_kernel(
    const float* __restrict__ src, _Float16* __restrict__ dst, int K, int N) {
  const size_t idx = (size_t)blockIdx.x * blockDim.x + threadIdx.x;
  const size_t k = idx % (size_t)K;
  const size_t n = idx / (size_t)K;
  dst[n * (size_t)K + k] = (_Float16)src[k * (size_t)N + n];
}

// ---------------------------------------------------------------------------
// Fused LayerNorm (shared g/b) + cast to f16.  One block per token row.
// ---------------------------------------------------------------------------
__global__ __launch_bounds__(256) void ln_cast_kernel(
    const float* __restrict__ x, const float* __restrict__ g,
    const float* __restrict__ bta, _Float16* __restrict__ out) {
  const size_t row = blockIdx.x;
  const float* xr = x + row * D_MODEL;
  const int tid  = threadIdx.x;
  const int lane = tid & 31;
  const int wave = tid >> 5;

  const float4 xv = ((const float4*)xr)[tid];   // 4 elems per thread, D=1024
  float v[4] = {xv.x, xv.y, xv.z, xv.w};
  float s = v[0] + v[1] + v[2] + v[3];
  float s2 = v[0]*v[0] + v[1]*v[1] + v[2]*v[2] + v[3]*v[3];
#pragma unroll
  for (int m = 1; m < 32; m <<= 1) {
    s  += __shfl_xor(s,  m, 32);
    s2 += __shfl_xor(s2, m, 32);
  }
  __shared__ float ws_s[8], ws_s2[8];
  if (lane == 0) { ws_s[wave] = s; ws_s2[wave] = s2; }
  __syncthreads();
  float S = 0.f, S2 = 0.f;
#pragma unroll
  for (int w = 0; w < 8; ++w) { S += ws_s[w]; S2 += ws_s2[w]; }
  const float mu  = S * (1.0f / D_MODEL);
  const float var = S2 * (1.0f / D_MODEL) - mu * mu;
  const float rs  = rsqrtf(var + 1e-5f);
  _Float16* orow = out + row * D_MODEL;
#pragma unroll
  for (int j = 0; j < 4; ++j) {
    const int col = tid * 4 + j;
    orow[col] = (_Float16)((v[j] - mu) * rs * g[col] + bta[col]);
  }
}

// ---------------------------------------------------------------------------
// WMMA GEMM: C[M,N] = A[M,K](f16, row-major) * Bt[N,K](f16, N-major) + bias.
// Block = 256 thr = 8 waves (4 M-rows x 2 N-cols), block tile 128x128,
// wave tile 32x64 (2x4 WMMA accumulators -> 8 WMMA / 12 b128 loads per
// k-step, ~21 FLOP/B from L2).  K-loop unrolled 2x so iteration i+1 loads
// issue under iteration i's WMMAs (hide the tail s_wait_loadcnt).
// ---------------------------------------------------------------------------
__global__ __launch_bounds__(256) void gemm_wmma_kernel(
    const _Float16* __restrict__ A, const _Float16* __restrict__ Bt,
    const float* __restrict__ bias, const float* __restrict__ resid,
    _Float16* __restrict__ outH, float* __restrict__ outF,
    int M, int N, int K, int mode) {
  const int lane = threadIdx.x & 31;
  const int wave = threadIdx.x >> 5;
  const int r    = lane & 15;
  const int hh   = (lane >> 4) & 1;
  const int m_base = blockIdx.y * 128 + (wave >> 1) * 32;   // 4 wave-rows
  const int n_base = blockIdx.x * 128 + (wave & 1)  * 64;   // 2 wave-cols
  (void)M;

  v8f c[2][4] = {};

  const _Float16* a_ptr0 = A + (size_t)m_base * K;
  const _Float16* a_ptr1 = A + (size_t)(m_base + 16) * K;
  // B fragment (contiguous k order, same as A): lane r holds column n+r, 16 contiguous K halfs
  // starting at k0 + 16*hh  -> one 32-byte contiguous load from Bt.
  const _Float16* b_ptr[4];
#pragma unroll
  for (int ni = 0; ni < 4; ++ni)
    b_ptr[ni] = Bt + (size_t)(n_base + 16 * ni + r) * K + 16 * hh;

#pragma unroll 2
  for (int k0 = 0; k0 < K; k0 += 32) {
    __builtin_prefetch(a_ptr0 + k0 + 256, 0, 1);   // global_prefetch_b8
    __builtin_prefetch(b_ptr[0] + k0 + 256, 0, 1);
    __builtin_prefetch(b_ptr[2] + k0 + 256, 0, 1);
    v16h af[2];
    af[0] = load_a16x32(a_ptr0 + k0, K, lane);
    af[1] = load_a16x32(a_ptr1 + k0, K, lane);
    v16h bf[4];
#pragma unroll
    for (int ni = 0; ni < 4; ++ni) bf[ni] = *(const v16h*)(b_ptr[ni] + k0);
#pragma unroll
    for (int mi = 0; mi < 2; ++mi)
#pragma unroll
      for (int ni = 0; ni < 4; ++ni)
        c[mi][ni] = wmma16x16x32(af[mi], bf[ni], c[mi][ni]);
  }

  // Epilogue. C layout: element i of v8f is (row = i + 8*hh, col = r).
#pragma unroll
  for (int mi = 0; mi < 2; ++mi)
#pragma unroll
    for (int ni = 0; ni < 4; ++ni)
#pragma unroll
      for (int i = 0; i < 8; ++i) {
        const int row = m_base + 16 * mi + i + 8 * hh;
        const int col = n_base + 16 * ni + r;
        float v = c[mi][ni][i] + bias[col];
        const size_t idx = (size_t)row * N + col;
        if (mode == EP_GELU) {
          v = 0.5f * v * (1.0f + erff(v * 0.70710678118654752f));
          outH[idx] = (_Float16)v;
        } else if (mode == EP_RESF32) {
          outF[idx] = v + resid[idx];
        } else {
          outH[idx] = (_Float16)v;
        }
      }
}

// ---------------------------------------------------------------------------
// Flash attention, one (b,h) per blockIdx.x, 128 queries per block (16/wave).
// No 1/sqrt(Dh) scaling (faithful to reference).  Output written in raw
// [B,H,N,Dh] flat order == reference's reshape to [B,N,D].
// ---------------------------------------------------------------------------
__global__ __launch_bounds__(256) void attn_flash_kernel(
    const _Float16* __restrict__ qkv, float* __restrict__ o) {
  const int b    = blockIdx.x / NHEAD;
  const int h    = blockIdx.x % NHEAD;
  const int lane = threadIdx.x & 31;
  const int wave = threadIdx.x >> 5;
  const int r    = lane & 15;
  const int hh   = (lane >> 4) & 1;
  const int q0   = blockIdx.y * 128 + wave * 16;   // wave's 16 query rows

  __shared__ __attribute__((aligned(32))) _Float16 lds_k[32 * 64];       // [key][dh]
  __shared__ __attribute__((aligned(32))) _Float16 lds_vt[64 * 32];      // [dh][key]
  __shared__ __attribute__((aligned(32))) _Float16 lds_p[8][16 * 32];    // per-wave P

  // Q fragments (rows q0..q0+15, Dh split 0..31 / 32..63), kept in VGPRs.
  const _Float16* qbase =
      qkv + ((size_t)(b * SEQ + q0)) * THREE_D + h * DH;
  const v16h aq0 = load_a16x32(qbase,      THREE_D, lane);
  const v16h aq1 = load_a16x32(qbase + 32, THREE_D, lane);

  float mrow[8], lrow[8];
  v8f oacc[4] = {};
#pragma unroll
  for (int i = 0; i < 8; ++i) { mrow[i] = -3.0e38f; lrow[i] = 0.f; }

  for (int kb = 0; kb < SEQ; kb += 32) {
    // --- cooperative stage of K (row-major) and V (transposed) into LDS ---
    {
      const int key = threadIdx.x >> 3;          // 0..31
      const int dh0 = (threadIdx.x & 7) * 8;     // 0..56, 8 halfs = 16B each
      const size_t grow = (size_t)(b * SEQ + kb + key) * THREE_D + h * DH + dh0;
#if USE_ASYNC_LDS
      // K tile: DMA straight into LDS, no VGPR bounce (tracked by ASYNCcnt).
      {
        unsigned lds_off = (unsigned)(uintptr_t)(&lds_k[key * 64 + dh0]);
        unsigned long long ga =
            (unsigned long long)(uintptr_t)(qkv + grow + D_MODEL);
        asm volatile("global_load_async_to_lds_b128 %0, %1, off"
                     :: "v"(lds_off), "v"(ga) : "memory");
      }
#else
      *(v8h*)(&lds_k[key * 64 + dh0]) = *(const v8h*)(qkv + grow + D_MODEL);
#endif
      // V tile: transposed scatter, must go through VGPRs.
      v8h vv = *(const v8h*)(qkv + grow + 2 * D_MODEL);
#pragma unroll
      for (int j = 0; j < 8; ++j) lds_vt[(dh0 + j) * 32 + key] = vv[j];
    }
#if USE_ASYNC_LDS
    asm volatile("s_wait_asynccnt 0" ::: "memory");
#endif
    __syncthreads();

    // --- S = Q * K^T for this 16(query) x 32(key) slab ------------------
    v8f s[2] = {};
#pragma unroll
    for (int t = 0; t < 2; ++t) {
      // B fragment: lane r = key (16*t + r), 16 contiguous dh per half-wave
      v16h bk0 = *(const v16h*)(&lds_k[(16 * t + r) * 64 +  0 + 16 * hh]);
      v16h bk1 = *(const v16h*)(&lds_k[(16 * t + r) * 64 + 32 + 16 * hh]);
      s[t] = wmma16x16x32(aq0, bk0, s[t]);
      s[t] = wmma16x16x32(aq1, bk1, s[t]);
    }

    // --- online softmax (per row = i + 8*hh; reduce across r via shfl) ---
    float pr[2][8];
#pragma unroll
    for (int i = 0; i < 8; ++i) {
      float mx = fmaxf(s[0][i], s[1][i]);
#pragma unroll
      for (int m = 1; m < 16; m <<= 1) mx = fmaxf(mx, __shfl_xor(mx, m, 32));
      const float mnew  = fmaxf(mrow[i], mx);
      const float scale = expf(mrow[i] - mnew);
      pr[0][i] = expf(s[0][i] - mnew);
      pr[1][i] = expf(s[1][i] - mnew);
      float ps = pr[0][i] + pr[1][i];
#pragma unroll
      for (int m = 1; m < 16; m <<= 1) ps += __shfl_xor(ps, m, 32);
      lrow[i] = lrow[i] * scale + ps;
      mrow[i] = mnew;
#pragma unroll
      for (int nt = 0; nt < 4; ++nt) oacc[nt][i] *= scale;
    }

    // --- C-layout -> A-layout for P via per-wave LDS round trip ---------
    _Float16* pw = &lds_p[wave][0];
#pragma unroll
    for (int t = 0; t < 2; ++t)
#pragma unroll
      for (int i = 0; i < 8; ++i)
        pw[(i + 8 * hh) * 32 + 16 * t + r] = (_Float16)pr[t][i];
    asm volatile("s_wait_dscnt 0" ::: "memory");   // wave-local LDS RAW fence
    const v16h ap = load_a16x32(pw, 32, lane);

    // --- O += P * V  (N tiles over Dh, B from transposed V in LDS) ------
#pragma unroll
    for (int nt = 0; nt < 4; ++nt) {
      v16h bv = *(const v16h*)(&lds_vt[(16 * nt + r) * 32 + 16 * hh]);
      oacc[nt] = wmma16x16x32(ap, bv, oacc[nt]);
    }
    __syncthreads();   // protect lds_k / lds_vt before next slab
  }

  // --- normalize and store in raw [B,H,N,Dh] flat order -----------------
  const size_t obase = ((size_t)(b * NHEAD + h)) * SEQ * DH;
#pragma unroll
  for (int i = 0; i < 8; ++i) {
    const float inv = 1.0f / lrow[i];
    const int row = q0 + i + 8 * hh;
#pragma unroll
    for (int nt = 0; nt < 4; ++nt)
      o[obase + (size_t)row * DH + 16 * nt + r] = oacc[nt][i] * inv;
  }
}

// ---------------------------------------------------------------------------
__global__ __launch_bounds__(256) void residual_add_kernel(
    const float* __restrict__ a, const float* __restrict__ b,
    float* __restrict__ out) {
  const size_t i = (size_t)blockIdx.x * blockDim.x + threadIdx.x;
  out[i] = a[i] + b[i];
}

// ---------------------------------------------------------------------------
extern "C" void kernel_launch(void* const* d_in, const int* in_sizes, int n_in,
                              void* d_out, int out_size, void* d_ws, size_t ws_size,
                              hipStream_t stream) {
  (void)in_sizes; (void)n_in; (void)out_size; (void)ws_size;
  const float* x     = (const float*)d_in[0];
  const float* w_qkv = (const float*)d_in[1];
  const float* b_qkv = (const float*)d_in[2];
  const float* ln_g  = (const float*)d_in[3];
  const float* ln_b  = (const float*)d_in[4];
  const float* w1    = (const float*)d_in[5];
  const float* b1    = (const float*)d_in[6];
  const float* w2    = (const float*)d_in[7];
  const float* b2    = (const float*)d_in[8];
  float* out = (float*)d_out;

  char* ws = (char*)d_ws;
  const size_t MB = 1024 * 1024;
  _Float16* wqkvt = (_Float16*)(ws + 0 * MB);   //  6 MB  [3072,1024] f16
  _Float16* w1t   = (_Float16*)(ws + 6 * MB);   //  8 MB  [4096,1024] f16
  _Float16* w2t   = (_Float16*)(ws + 14 * MB);  //  8 MB  [1024,4096] f16
  _Float16* xn_h  = (_Float16*)(ws + 22 * MB);  //  8 MB  [4096,1024] f16 (reused for xn2)
  _Float16* qkv_h = (_Float16*)(ws + 30 * MB);  // 24 MB  [4096,3072] f16
  float*    attn  = (float*)   (ws + 54 * MB);  // 16 MB  [B,H,N,Dh]  f32
  float*    x2    = (float*)   (ws + 70 * MB);  // 16 MB  [4096,1024] f32
  _Float16* h_h   = (_Float16*)(ws + 30 * MB);  // 32 MB, reuses qkv_h+attn (both dead)

  // 1) Weights -> f16, transposed to N-major (one-time, bandwidth-trivial).
  transpose_cast_kernel<<<(D_MODEL * THREE_D) / 256, 256, 0, stream>>>(w_qkv, wqkvt, D_MODEL, THREE_D);
  transpose_cast_kernel<<<(D_MODEL * HID)     / 256, 256, 0, stream>>>(w1,    w1t,   D_MODEL, HID);
  transpose_cast_kernel<<<(HID * D_MODEL)     / 256, 256, 0, stream>>>(w2,    w2t,   HID,     D_MODEL);

  // 2) xn = LN(x) -> f16
  ln_cast_kernel<<<ROWS, 256, 0, stream>>>(x, ln_g, ln_b, xn_h);

  // 3) qkv = xn @ w_qkv + b_qkv  (f16 out)
  gemm_wmma_kernel<<<dim3(THREE_D / 128, ROWS / 128), 256, 0, stream>>>(
      xn_h, wqkvt, b_qkv, nullptr, qkv_h, nullptr, ROWS, THREE_D, D_MODEL, EP_F16);

  // 4) flash attention (no Dh scaling), raw-reshape output order
  attn_flash_kernel<<<dim3(BATCH * NHEAD, SEQ / 128), 256, 0, stream>>>(qkv_h, attn);

  // 5) x2 = x + attn_out
  residual_add_kernel<<<(ROWS * D_MODEL) / 256, 256, 0, stream>>>(x, attn, x2);

  // 6) xn2 = LN(x2) -> f16 (reuse buffer)
  ln_cast_kernel<<<ROWS, 256, 0, stream>>>(x2, ln_g, ln_b, xn_h);

  // 7) h = gelu(xn2 @ w1 + b1) -> f16
  gemm_wmma_kernel<<<dim3(HID / 128, ROWS / 128), 256, 0, stream>>>(
      xn_h, w1t, b1, nullptr, h_h, nullptr, ROWS, HID, D_MODEL, EP_GELU);

  // 8) out = x2 + h @ w2 + b2  (f32)
  gemm_wmma_kernel<<<dim3(D_MODEL / 128, ROWS / 128), 256, 0, stream>>>(
      h_h, w2t, b2, x2, nullptr, out, ROWS, D_MODEL, HID, EP_RESF32);
}
